// HiddenCDE_50311246905432
// MI455X (gfx1250) — matrix operand order repaired and checked by reading, not compiled, not run on hardware
//
#include <hip/hip_runtime.h>
#include <hip/hip_bf16.h>

// ---------------------------------------------------------------------------
// Neural CDE on gfx1250: 16 cooperating workgroups (one WGP each).
//  - W1/W2 (128x128) matvecs: 8 waves x 16-row tiles, v_wmma_f32_16x16x32_bf16
//  - W3 (4096x128 / 2048x128): sliced across the 16 WGPs, WMMA per tile
//  - per f-eval: each WGP publishes 8 floats of f; device-scope barrier; all
//    WGPs redundantly advance identical dopri5 state (bit-identical FP).
// ---------------------------------------------------------------------------

typedef __bf16 v16bf __attribute__((ext_vector_type(16)));
typedef __bf16 bf8v  __attribute__((ext_vector_type(8)));
typedef float  v8f   __attribute__((ext_vector_type(8)));

#define NWG   16
#define TPB   256
#define NSTEP 1023
#define L_PAST 1024

// dopri5 a_ij, packed rows: st1@0, st2@1, st3@3, st4@6, st5@10
__constant__ float c_A[15] = {
  0.2f,
  3.0f/40.0f, 9.0f/40.0f,
  44.0f/45.0f, -56.0f/15.0f, 32.0f/9.0f,
  19372.0f/6561.0f, -25360.0f/2187.0f, 64448.0f/6561.0f, -212.0f/729.0f,
  9017.0f/3168.0f, -355.0f/33.0f, 46732.0f/5247.0f, 49.0f/176.0f, -5103.0f/18656.0f
};

__device__ __forceinline__ float softplus_f(float x) {
  return fmaxf(x, 0.0f) + log1pf(expf(-fabsf(x)));
}

// 16 output rows [row0, row0+16) of W(row-major, K=128, bf16) @ x (LDS f32[128])
// A = weight tile (16x32), B = x broadcast across the 16 columns.
__device__ __forceinline__ v8f wmma_mv_tile(const __bf16* __restrict__ W,
                                            int row0,
                                            const float* __restrict__ x) {
  const int lane = threadIdx.x & 31;
  const int m    = row0 + (lane & 15);
  const int base = (lane & 16) ? 8 : 0;   // A layout: hi half-wave holds K+8
  v8f acc = {0.f,0.f,0.f,0.f,0.f,0.f,0.f,0.f};
#pragma unroll
  for (int kb = 0; kb < 128; kb += 32) {
    const __bf16* wr = W + (size_t)m * 128 + kb + base;
    bf8v lo = *(const bf8v*)(wr);        // K = base..base+7
    bf8v hi = *(const bf8v*)(wr + 16);   // K = base+16..base+23
    v16bf a = __builtin_shufflevector(lo, hi, 0,1,2,3,4,5,6,7,8,9,10,11,12,13,14,15);
    const float* xr = x + kb + base;     // B: same K order as A, constant across N
    v16bf b = {(__bf16)xr[0],(__bf16)xr[1],(__bf16)xr[2],(__bf16)xr[3],(__bf16)xr[4],(__bf16)xr[5],(__bf16)xr[6],(__bf16)xr[7],(__bf16)xr[16],(__bf16)xr[17],(__bf16)xr[18],(__bf16)xr[19],(__bf16)xr[20],(__bf16)xr[21],(__bf16)xr[22],(__bf16)xr[23]};
    acc = __builtin_amdgcn_wmma_f32_16x16x32_bf16(false, a, false, b,
                                                  (short)0, acc, false, false);
  }
  return acc;
}

// Device-scope sense-reversing barrier across NWG blocks.
__device__ __forceinline__ void grid_barrier(unsigned* bar, unsigned* epoch) {
  __threadfence();            // release this thread's prior global stores
  __syncthreads();
  if (threadIdx.x == 0) {
    const unsigned e = ++(*epoch);
    unsigned prev = __hip_atomic_fetch_add(&bar[0], 1u, __ATOMIC_ACQ_REL,
                                           __HIP_MEMORY_SCOPE_AGENT);
    if (prev == NWG - 1) {
      __hip_atomic_store(&bar[0], 0u, __ATOMIC_RELAXED, __HIP_MEMORY_SCOPE_AGENT);
      __hip_atomic_fetch_add(&bar[1], 1u, __ATOMIC_RELEASE, __HIP_MEMORY_SCOPE_AGENT);
    } else {
      while (__hip_atomic_load(&bar[1], __ATOMIC_ACQUIRE,
                               __HIP_MEMORY_SCOPE_AGENT) < e) {
        __builtin_amdgcn_s_sleep(2);
      }
    }
  }
  __syncthreads();
}

// ---------------------------- setup kernels --------------------------------

__global__ void cde_f2bf(const float* __restrict__ s, __bf16* __restrict__ d, int n) {
  int i = blockIdx.x * blockDim.x + threadIdx.x;
  if (i < n) d[i] = (__bf16)s[i];
}

__global__ void cde_zero_bar(unsigned* bar) {
  if (threadIdx.x < 2) bar[threadIdx.x] = 0u;
}

__global__ void cde_build_dx_past(const float* __restrict__ y_past,
                                  const float* __restrict__ cx,
                                  const float* __restrict__ t,
                                  float* __restrict__ dxp) {
  const int i = blockIdx.x;          // 0..1022
  const int c = threadIdx.x;         // 0..31
  const float dt = t[i+1] - t[i];
  float num;
  if (c < 16)       num = y_past[(i+1)*16 + c]      - y_past[i*16 + c];
  else if (c < 31)  num = cx[(i+1)*15 + (c-16)]     - cx[i*15 + (c-16)];
  else              num = dt;
  dxp[i*32 + c] = num / dt;
}

__global__ void cde_build_dx_fut(const float* __restrict__ cx,
                                 const float* __restrict__ t,
                                 float* __restrict__ dxf) {
  const int i = blockIdx.x;          // 0..1022
  const int c = threadIdx.x;         // 0..15
  const int a = L_PAST + i, b = L_PAST + i + 1;
  const float dt = t[b] - t[a];
  const float num = (c < 15) ? (cx[b*15 + c] - cx[a*15 + c]) : dt;
  dxf[i*16 + c] = num / dt;
}

__global__ void cde_init_z0(const float* __restrict__ iW, const float* __restrict__ ib,
                            const float* __restrict__ y_past, const float* __restrict__ cx,
                            const float* __restrict__ t, float* __restrict__ z0) {
  const int r = threadIdx.x;         // 0..127
  float acc = ib[r];
  for (int c = 0; c < 32; ++c) {
    float xc;
    if (c < 16)      xc = y_past[c];
    else if (c < 31) xc = cx[c - 16];
    else             xc = t[0];
    acc = fmaf(iW[r*32 + c], xc, acc);
  }
  z0[r] = acc;
}

// ----------------------------- main kernel ---------------------------------

__global__ __launch_bounds__(TPB) void cde_main(
    const __bf16* __restrict__ w1p, const __bf16* __restrict__ w2p, const __bf16* __restrict__ w3p,
    const __bf16* __restrict__ w1f, const __bf16* __restrict__ w2f, const __bf16* __restrict__ w3f,
    const float* __restrict__ b1p, const float* __restrict__ b2p, const float* __restrict__ b3p,
    const float* __restrict__ b1f, const float* __restrict__ b2f, const float* __restrict__ b3f,
    const float* __restrict__ z0,  const float* __restrict__ dxp, const float* __restrict__ dxf,
    const float* __restrict__ tarr, const float* __restrict__ roW, const float* __restrict__ rob,
    float* fbuf, unsigned* bar, float* __restrict__ out)
{
  __shared__ float zcur[128], zin[128], h1[128], h2[128];
  __shared__ float karr[6][128];
  __shared__ float gbuf[256];
  __shared__ float dxs[32];

  const int tid  = threadIdx.x;
  const int wg   = blockIdx.x;
  const int wave = tid >> 5;
  unsigned epoch = 0;

  if (tid < 128) zcur[tid] = z0[tid];
  __syncthreads();

  for (int phase = 0; phase < 2; ++phase) {
    const __bf16* W1 = phase ? w1f : w1p;
    const __bf16* W2 = phase ? w2f : w2p;
    const __bf16* W3 = phase ? w3f : w3p;
    const float*  B1 = phase ? b1f : b1p;
    const float*  B2 = phase ? b2f : b2p;
    const float*  B3 = phase ? b3f : b3p;
    const float*  dX = phase ? dxf : dxp;
    const float*  tt = phase ? (tarr + L_PAST) : tarr;
    const int ctrl    = phase ? 16 : 32;
    const int tilesWG = phase ? 8  : 16;    // W3 row-tiles per workgroup
    const int rowsWG  = tilesWG * 16;

    if (phase == 1 && wg == 0 && tid < 8) {  // output row 0 = readout(zL)
      float acc = rob[tid];
      for (int q = 0; q < 128; ++q) acc = fmaf(roW[tid*128 + q], zcur[q], acc);
      out[tid] = acc;
    }

    for (int s = 0; s < NSTEP; ++s) {
      if (tid < ctrl) dxs[tid] = dX[s*ctrl + tid];
      const float dt = tt[s+1] - tt[s];
      const float h  = 0.5f * dt;            // SUBSTEPS = 2

      for (int sub = 0; sub < 2; ++sub) {
        for (int st = 0; st < 6; ++st) {
          // stage input  z_st = z + h * sum_j a[st][j] * k_j   (thread-local in q)
          if (tid < 128) {
            float zz = zcur[tid];
            const int aoff = (st*(st-1)) >> 1;
            for (int j = 0; j < st; ++j)
              zz = fmaf(h * c_A[aoff + j], karr[j][tid], zz);
            zin[tid] = zz;
          }
          __syncthreads();

          // layer 1: h1 = softplus(W1 z + b1)   (8 waves x 16-row WMMA tiles)
          {
            v8f acc = wmma_mv_tile(W1, wave*16, zin);
            if ((tid & 15) == 0) {
              const int half = (tid >> 4) & 1;
#pragma unroll
              for (int r = 0; r < 8; ++r) {
                const int m = wave*16 + half*8 + r;
                h1[m] = softplus_f(acc[r] + B1[m]);
              }
            }
          }
          __syncthreads();

          // layer 2: h2 = softplus(W2 h1 + b2)
          {
            v8f acc = wmma_mv_tile(W2, wave*16, h1);
            if ((tid & 15) == 0) {
              const int half = (tid >> 4) & 1;
#pragma unroll
              for (int r = 0; r < 8; ++r) {
                const int m = wave*16 + half*8 + r;
                h2[m] = softplus_f(acc[r] + B2[m]);
              }
            }
          }
          __syncthreads();

          // layer 3 slice: g = tanh(W3[rows of this WG] h2 + b3)
          for (int tile = wave; tile < tilesWG; tile += 8) {
            const int grow0 = wg*rowsWG + tile*16;
            v8f acc = wmma_mv_tile(W3, grow0, h2);
            if ((tid & 15) == 0) {
              const int half = (tid >> 4) & 1;
#pragma unroll
              for (int r = 0; r < 8; ++r) {
                const int lr = tile*16 + half*8 + r;
                gbuf[lr] = tanhf(acc[r] + B3[grow0 + half*8 + r]);
              }
            }
          }
          __syncthreads();

          // contraction: this WG's 8 components of f = G_slice @ dxdt
          if (tid < 8) {
            float fv = 0.0f;
            const float* gr = &gbuf[tid*ctrl];
            for (int c = 0; c < ctrl; ++c) fv = fmaf(gr[c], dxs[c], fv);
            __hip_atomic_store(&fbuf[st*128 + wg*8 + tid], fv,
                               __ATOMIC_RELAXED, __HIP_MEMORY_SCOPE_AGENT);
          }

          grid_barrier(bar, &epoch);

          // gather full k_st (agent-scope loads bypass L0 -> fresh from L2)
          if (tid < 128)
            karr[st][tid] = __hip_atomic_load(&fbuf[st*128 + tid],
                                              __ATOMIC_RELAXED, __HIP_MEMORY_SCOPE_AGENT);
        } // stages

        if (tid < 128) {
          const float kz = (35.0f/384.0f)    * karr[0][tid]
                         + (500.0f/1113.0f)  * karr[2][tid]
                         + (125.0f/192.0f)   * karr[3][tid]
                         - (2187.0f/6784.0f) * karr[4][tid]
                         + (11.0f/84.0f)     * karr[5][tid];
          zcur[tid] += h * kz;
        }
        __syncthreads();
      } // substeps

      if (phase == 1 && wg == 0 && tid < 8) {  // output row s+1 = readout(z)
        float acc = rob[tid];
        for (int q = 0; q < 128; ++q) acc = fmaf(roW[tid*128 + q], zcur[q], acc);
        out[(s+1)*8 + tid] = acc;
      }
    } // steps
    __syncthreads();
  } // phase
}

// ------------------------------ launcher -----------------------------------

extern "C" void kernel_launch(void* const* d_in, const int* in_sizes, int n_in,
                              void* d_out, int out_size, void* d_ws, size_t ws_size,
                              hipStream_t stream) {
  const float* y_past = (const float*)d_in[0];
  const float* t      = (const float*)d_in[1];
  const float* cx     = (const float*)d_in[2];
  // d_in[3] = input_length (int, == 1024, compile-time constant here)
  const float* initW  = (const float*)d_in[4];
  const float* initb  = (const float*)d_in[5];
  const float* pW1 = (const float*)d_in[6];  const float* pb1 = (const float*)d_in[7];
  const float* pW2 = (const float*)d_in[8];  const float* pb2 = (const float*)d_in[9];
  const float* pW3 = (const float*)d_in[10]; const float* pb3 = (const float*)d_in[11];
  const float* fW1 = (const float*)d_in[12]; const float* fb1 = (const float*)d_in[13];
  const float* fW2 = (const float*)d_in[14]; const float* fb2 = (const float*)d_in[15];
  const float* fW3 = (const float*)d_in[16]; const float* fb3 = (const float*)d_in[17];
  const float* roW = (const float*)d_in[18]; const float* rob = (const float*)d_in[19];
  float* out = (float*)d_out;

  char* ws = (char*)d_ws;
  size_t off = 0;
  auto take = [&](size_t bytes) -> char* {
    char* p = ws + off;
    off = (off + bytes + 255) & ~(size_t)255;
    return p;
  };
  unsigned* bar  = (unsigned*)take(2 * sizeof(unsigned));
  float*    fbuf = (float*)take(6 * 128 * sizeof(float));
  float*    z0   = (float*)take(128 * sizeof(float));
  float*    dxp  = (float*)take((size_t)NSTEP * 32 * sizeof(float));
  float*    dxf  = (float*)take((size_t)NSTEP * 16 * sizeof(float));
  __bf16*   w1p  = (__bf16*)take(16384 * 2);
  __bf16*   w2p  = (__bf16*)take(16384 * 2);
  __bf16*   w3p  = (__bf16*)take((size_t)524288 * 2);
  __bf16*   w1f  = (__bf16*)take(16384 * 2);
  __bf16*   w2f  = (__bf16*)take(16384 * 2);
  __bf16*   w3f  = (__bf16*)take((size_t)262144 * 2);

  cde_zero_bar<<<1, 32, 0, stream>>>(bar);
  cde_f2bf<<<(16384 + 255) / 256, 256, 0, stream>>>(pW1, w1p, 16384);
  cde_f2bf<<<(16384 + 255) / 256, 256, 0, stream>>>(pW2, w2p, 16384);
  cde_f2bf<<<(524288 + 255) / 256, 256, 0, stream>>>(pW3, w3p, 524288);
  cde_f2bf<<<(16384 + 255) / 256, 256, 0, stream>>>(fW1, w1f, 16384);
  cde_f2bf<<<(16384 + 255) / 256, 256, 0, stream>>>(fW2, w2f, 16384);
  cde_f2bf<<<(262144 + 255) / 256, 256, 0, stream>>>(fW3, w3f, 262144);
  cde_build_dx_past<<<NSTEP, 32, 0, stream>>>(y_past, cx, t, dxp);
  cde_build_dx_fut<<<NSTEP, 16, 0, stream>>>(cx, t, dxf);
  cde_init_z0<<<1, 128, 0, stream>>>(initW, initb, y_past, cx, t, z0);

  cde_main<<<NWG, TPB, 0, stream>>>(w1p, w2p, w3p, w1f, w2f, w3f,
                                    pb1, pb2, pb3, fb1, fb2, fb3,
                                    z0, dxp, dxf, t, roW, rob,
                                    fbuf, bar, out);
}
